// LinearLoopLayer_21251498180722
// MI455X (gfx1250) — matrix operand order repaired and checked by reading, not compiled, not run on hardware
//
#include <hip/hip_runtime.h>
#include <stdint.h>

// ---------------------------------------------------------------------------
// out[4096,4096] = x[4096,4096] @ W[4096,4096]^T + b[4096]   (all fp32)
//
// bf16 split-precision GEMM on the CDNA5 WMMA path:
//   f32 v -> hi = bf16(v), lo = bf16(v - float(hi))
//   acc += hi_a*hi_b + hi_a*lo_b + lo_a*hi_b      (f32 accumulators)
//
// This round: stage the next K-tile with GLOBAL_LOAD_ASYNC_TO_LDS_B128
// (ASYNCcnt-tracked, zero destination VGPRs) instead of holding 48 float4
// registers across the MAC phase. This pulls the kernel back under the
// 256-VGPR addressable window (eliminating the s_set_vgpr_msb storm seen in
// round 3) while keeping the 64x64-per-wave tile (48 WMMA / wave-iter).
// ---------------------------------------------------------------------------

typedef __attribute__((ext_vector_type(16))) __bf16 v16bf;
typedef __attribute__((ext_vector_type(8)))  __bf16 v8bf;
typedef __attribute__((ext_vector_type(8)))  float  v8f;

#define MAT_N  4096
#define BM     256
#define BN     128
#define BK     32
#define PITCH  40            // bf16 per LDS row (32 + 8 pad), rows 16B-aligned
#define NIT    (MAT_N / BK)
#define SF_A   (BM * BK)     // floats in the A part of the fp32 staging buffer

// Async global->LDS copy, 16 bytes per lane. VDST = LDS byte offset (low 32
// bits of the flat shared address), VADDR = 64-bit global address.
__device__ __forceinline__ void async_ld_b128(uint32_t lds_off, const float* g) {
    asm volatile("global_load_async_to_lds_b128 %0, %1, off"
                 :: "v"(lds_off), "v"(g) : "memory");
}
__device__ __forceinline__ void wait_async0() {
    asm volatile("s_wait_asynccnt 0" ::: "memory");
}

// ---- split 16 f32 (4x float4) into bf16 hi/lo, store to an LDS tile row ----
__device__ __forceinline__ void split_store(__bf16* __restrict__ hiRow,
                                            __bf16* __restrict__ loRow,
                                            int col, const float4* v) {
    const float* f = (const float*)v;
#pragma unroll
    for (int c = 0; c < 2; ++c) {            // two 8-element (16B) groups
        v8bf h, l;
#pragma unroll
        for (int e = 0; e < 8; ++e) {
            float  x  = f[c * 8 + e];
            __bf16 hh = (__bf16)x;           // v_cvt_pk_bf16_f32 (paired)
            h[e] = hh;
            l[e] = (__bf16)(x - (float)hh);  // v_fma_mix residual
        }
        *(v8bf*)(hiRow + col + c * 8) = h;   // ds_store_b128
        *(v8bf*)(loRow + col + c * 8) = l;   // ds_store_b128
    }
}

// ---- A fragment 16x32 bf16 (ISA 7.12.2):
//  lane m    : bf16[0..7]=K 0..7,  bf16[8..15]=K 16..23
//  lane m+16 : bf16[0..7]=K 8..15, bf16[8..15]=K 24..31
__device__ __forceinline__ v16bf frag_a(const __bf16* __restrict__ row, int khalf) {
    v8bf x0 = *(const v8bf*)(row + khalf * 8);
    v8bf x1 = *(const v8bf*)(row + 16 + khalf * 8);
    return __builtin_shufflevector(x0, x1, 0, 1, 2, 3, 4, 5, 6, 7,
                                           8, 9, 10, 11, 12, 13, 14, 15);
}

// ---- B fragment 32x16 bf16: lanes 0..15 K 0..7,16..23; lanes 16..31 K 8..15,24..31 ----
__device__ __forceinline__ v16bf frag_b(const __bf16* __restrict__ row, int khalf) {
    v8bf x0 = *(const v8bf*)(row + khalf * 8);
    v8bf x1 = *(const v8bf*)(row + 16 + khalf * 8);
    return __builtin_shufflevector(x0, x1, 0, 1, 2, 3, 4, 5, 6, 7,
                                           8, 9, 10, 11, 12, 13, 14, 15);
}

extern "C" __global__ __launch_bounds__(256)
void linear_wmma_bf16x3_kernel(const float* __restrict__ X,
                               const float* __restrict__ W,
                               const float* __restrict__ bias,
                               float* __restrict__ out) {
    // bf16 hi/lo tiles, double-buffered: 120 KB
    __shared__ __bf16 Ah[2][BM][PITCH];
    __shared__ __bf16 Al[2][BM][PITCH];
    __shared__ __bf16 Bh[2][BN][PITCH];
    __shared__ __bf16 Bl[2][BN][PITCH];
    // raw fp32 staging for the in-flight tile (async target): 48 KB
    __shared__ float Sf[(BM + BN) * BK];

    const int tid   = threadIdx.x;
    const int lane  = tid & 31;
    const int wave  = tid >> 5;      // 0..7
    const int waveM = wave >> 1;     // 0..3 -> 64-row strip
    const int waveN = wave & 1;      // 0..1 -> 64-col strip

    const int blockM = blockIdx.y * BM;
    const int blockN = blockIdx.x * BN;

    const int lm    = lane & 15;
    const int khalf = lane >> 4;

    // stager mapping: A = 256 rows x 32 f32 (rows ldRow and ldRow+128),
    // B = 128 rows x 32 f32; each thread owns three 16-float segments.
    const int ldRow = tid >> 1;          // 0..127
    const int ldCol = (tid & 1) * 16;    // 0 or 16

    const float* gA0 = X + (size_t)(blockM + ldRow) * MAT_N + ldCol;
    const float* gA1 = gA0 + (size_t)128 * MAT_N;
    const float* gB  = W + (size_t)(blockN + ldRow) * MAT_N + ldCol;

    // This thread's staging segments (generic ptrs for ds reads, low 32 bits
    // of the flat shared address = LDS byte offset for the async engine).
    float* sA0 = &Sf[ldRow * BK + ldCol];
    float* sA1 = &Sf[(128 + ldRow) * BK + ldCol];
    float* sB  = &Sf[SF_A + ldRow * BK + ldCol];
    const uint32_t oA0 = (uint32_t)(uintptr_t)sA0;
    const uint32_t oA1 = (uint32_t)(uintptr_t)sA1;
    const uint32_t oB  = (uint32_t)(uintptr_t)sB;

    v8f acc[4][4];
#pragma unroll
    for (int i = 0; i < 4; ++i)
#pragma unroll
        for (int j = 0; j < 4; ++j)
            acc[i][j] = (v8f){0.f, 0.f, 0.f, 0.f, 0.f, 0.f, 0.f, 0.f};

    // ---- prologue: stage tile 0 via ordinary loads into buffer 0 ----
    {
        float4 ra0[4], ra1[4], rb[4];
#pragma unroll
        for (int v = 0; v < 4; ++v) {
            ra0[v] = ((const float4*)gA0)[v];
            ra1[v] = ((const float4*)gA1)[v];
            rb[v]  = ((const float4*)gB)[v];
        }
        split_store(&Ah[0][ldRow][0],       &Al[0][ldRow][0],       ldCol, ra0);
        split_store(&Ah[0][128 + ldRow][0], &Al[0][128 + ldRow][0], ldCol, ra1);
        split_store(&Bh[0][ldRow][0],       &Bl[0][ldRow][0],       ldCol, rb);
    }
    __syncthreads();

    for (int it = 0; it < NIT; ++it) {
        const int cur = it & 1;
        const int nxt = cur ^ 1;

        // ---- issue async global->LDS copies for tile it+1 (no VGPRs held)
        if (it + 1 < NIT) {
            const int kNext = (it + 1) * BK;
#pragma unroll
            for (int v = 0; v < 4; ++v) {
                async_ld_b128(oA0 + v * 16, gA0 + kNext + v * 4);
                async_ld_b128(oA1 + v * 16, gA1 + kNext + v * 4);
                async_ld_b128(oB  + v * 16, gB  + kNext + v * 4);
            }
            if (it + 2 < NIT) {   // warm L2 for the following async batch
                __builtin_prefetch(gA0 + kNext + BK, 0, 3);
                __builtin_prefetch(gA1 + kNext + BK, 0, 3);
                __builtin_prefetch(gB + kNext + BK, 0, 3);
            }
        }

        // ---- consume tile `it`: B fragments resident, A fragments transient
        v16bf bHi[4], bLo[4];
#pragma unroll
        for (int j = 0; j < 4; ++j) {
            const int r = waveN * 64 + j * 16 + lm;
            bHi[j] = frag_b(&Bh[cur][r][0], khalf);
            bLo[j] = frag_b(&Bl[cur][r][0], khalf);
        }
#pragma unroll
        for (int i = 0; i < 4; ++i) {
            const int r = waveM * 64 + i * 16 + lm;
            const v16bf aHi = frag_a(&Ah[cur][r][0], khalf);
            const v16bf aLo = frag_a(&Al[cur][r][0], khalf);
#pragma unroll
            for (int j = 0; j < 4; ++j) {
                acc[i][j] = __builtin_amdgcn_wmma_f32_16x16x32_bf16(
                    false, aHi, false, bHi[j], (short)0, acc[i][j], false, false);
                acc[i][j] = __builtin_amdgcn_wmma_f32_16x16x32_bf16(
                    false, aHi, false, bLo[j], (short)0, acc[i][j], false, false);
                acc[i][j] = __builtin_amdgcn_wmma_f32_16x16x32_bf16(
                    false, aLo, false, bHi[j], (short)0, acc[i][j], false, false);
            }
        }

        // ---- drain async, convert own staged fp32 -> bf16 hi/lo buffers ----
        if (it + 1 < NIT) {
            wait_async0();
            float4 ra0[4], ra1[4], rb[4];
#pragma unroll
            for (int v = 0; v < 4; ++v) {
                ra0[v] = ((const float4*)sA0)[v];   // ds_load_b128 (own data)
                ra1[v] = ((const float4*)sA1)[v];
                rb[v]  = ((const float4*)sB)[v];
            }
            split_store(&Ah[nxt][ldRow][0],       &Al[nxt][ldRow][0],       ldCol, ra0);
            split_store(&Ah[nxt][128 + ldRow][0], &Al[nxt][128 + ldRow][0], ldCol, ra1);
            split_store(&Bh[nxt][ldRow][0],       &Bl[nxt][ldRow][0],       ldCol, rb);
        }
        __syncthreads();
    }

    // ---- epilogue: C/D layout (VGPR r -> row r + 8*khalf, col = lane&15) ----
#pragma unroll
    for (int j = 0; j < 4; ++j) {
        const int col = blockN + waveN * 64 + j * 16 + lm;
        const float bv = bias[col];
#pragma unroll
        for (int i = 0; i < 4; ++i) {
            const int rowBase = blockM + waveM * 64 + i * 16 + khalf * 8;
#pragma unroll
            for (int r = 0; r < 8; ++r) {
                out[(size_t)(rowBase + r) * MAT_N + col] = acc[i][j][r] + bv;
            }
        }
    }
}

extern "C" void kernel_launch(void* const* d_in, const int* in_sizes, int n_in,
                              void* d_out, int out_size, void* d_ws, size_t ws_size,
                              hipStream_t stream) {
    (void)in_sizes; (void)n_in; (void)d_ws; (void)ws_size; (void)out_size;
    const float* x = (const float*)d_in[0];
    const float* w = (const float*)d_in[1];
    const float* b = (const float*)d_in[2];
    float* o = (float*)d_out;

    dim3 grid(MAT_N / BN, MAT_N / BM);   // 32 x 16 blocks
    dim3 block(256);                     // 8 wave32 waves
    linear_wmma_bf16x3_kernel<<<grid, block, 0, stream>>>(x, w, b, o);
}
